// AttentionNN_2018634629851
// MI455X (gfx1250) — matrix operand order repaired and checked by reading, not compiled, not run on hardware
//
#include <hip/hip_runtime.h>
#include <hip/hip_bf16.h>
#include <math.h>

// Problem dims (match reference)
#define BB 32
#define TT 64
#define EE 512
#define HH 1024
#define LL 2
#define VTOK 32000

typedef __bf16 bf16_t;
typedef __bf16 v16bf __attribute__((ext_vector_type(16)));
typedef __bf16 v8bf  __attribute__((ext_vector_type(8)));
typedef float  v8f   __attribute__((ext_vector_type(8)));

union BF16x16 { v16bf v; v8bf h[2]; };

// ---------------- CDNA5 async-copy helpers ----------------
// GLOBAL_LOAD_ASYNC_TO_LDS_B128 (GVS mode): mem = SADDR(64b SGPR) + VADDR(32b VGPR)
// dest LDS byte address in VDST VGPR.  Tracked by ASYNCcnt.
__device__ __forceinline__ void async_ld_b128(uint32_t lds_byte_off,
                                              uint32_t gl_byte_off,
                                              const void* base) {
  asm volatile("global_load_async_to_lds_b128 %0, %1, %2"
               :: "v"(lds_byte_off), "v"(gl_byte_off), "s"(base)
               : "memory");
}
__device__ __forceinline__ void wait_async0() {
  asm volatile("s_wait_asynccnt 0" ::: "memory");
}
// Low 32 bits of a generic pointer to LDS = LDS byte offset (aperture rule).
__device__ __forceinline__ uint32_t lds_addr(const void* p) {
  return (uint32_t)(uintptr_t)p;
}

// ---------------- WMMA fragment helpers (wave32, 16x16x32 bf16) ----------------
// A fragment, 16x32 (MxK), row-major source with leading dim lda.
__device__ __forceinline__ v16bf load_a_frag(const bf16_t* __restrict__ A, int lda,
                                             int m0, int k0, int lane) {
  const int row  = m0 + (lane & 15);
  const int koff = (lane >> 4) << 3;
  BF16x16 f;
  f.h[0] = *(const v8bf*)(A + (size_t)row * lda + (k0 + koff));
  f.h[1] = *(const v8bf*)(A + (size_t)row * lda + (k0 + 16 + koff));
  return f.v;
}

// B fragment, 32x16 (KxN), row-major source with leading dim ldb.
__device__ __forceinline__ v16bf load_b_frag(const bf16_t* __restrict__ Bm, int ldb,
                                             int k0, int n0, int lane) {
  const int col  = n0 + (lane & 15);
  const int koff = (lane >> 4) << 3;
  BF16x16 f;
#pragma unroll
  for (int s = 0; s < 8; ++s) {
    f.v[s]     = Bm[(size_t)(k0 + koff + s) * ldb + col];
    f.v[8 + s] = Bm[(size_t)(k0 + 16 + koff + s) * ldb + col];
  }
  return f.v;
}

__device__ __forceinline__ v8f wmma_bf16(v16bf a, v16bf b, v8f c) {
  return __builtin_amdgcn_wmma_f32_16x16x32_bf16(false, a, false, b, (short)0, c,
                                                 false, false);
}

__device__ __forceinline__ float sigmoidf_(float x) {
  return 1.0f / (1.0f + __expf(-x));
}

// ---------------- elementwise / gather kernels ----------------
__global__ void k_f32_to_bf16(const float* __restrict__ in, bf16_t* __restrict__ out,
                              long n) {
  long i = (long)blockIdx.x * blockDim.x + threadIdx.x;
  const long stride = (long)gridDim.x * blockDim.x;
  for (; i < n; i += stride) out[i] = (bf16_t)in[i];
}

// out[t*B + b, :] = emb[tok[b, t], :]   (bf16)
__global__ void k_gather_embed(const int* __restrict__ tok, const float* __restrict__ emb,
                               bf16_t* __restrict__ out) {
  const int i = blockIdx.x;         // t*B + b
  const int b = i % BB, t = i / BB;
  const int v = tok[b * TT + t];
  const float* src = emb + (size_t)v * EE;
  bf16_t* dst = out + (size_t)i * EE;
  for (int e = threadIdx.x; e < EE; e += blockDim.x) dst[e] = (bf16_t)src[e];
}

// ---------------- generic WMMA GEMM: C = A(bf16) @ B(bf16) + bias ----------------
// block = 4 waves (128 thr).  Wave w owns M-tile m0 = by*64 + w*16; all four waves
// share a 32(K)x64(N) B tile staged in LDS via async-to-LDS, double buffered.
// grid = (N/64, ceil(M/64)).
__global__ __launch_bounds__(128) void k_wmma_gemm(
    const bf16_t* __restrict__ A, const bf16_t* __restrict__ Bm,
    const float* __restrict__ bias, float* __restrict__ C,
    bf16_t* __restrict__ Cbf, int M, int N, int K) {
  __shared__ bf16_t sB[2][32][72];                 // +8 bf16 row pad (bank spread)
  const int tid  = threadIdx.x;
  const int lane = tid & 31;
  const int wv   = tid >> 5;
  const int n0   = blockIdx.x * 64;
  const int m0   = blockIdx.y * 64 + wv * 16;
  const bool active = (m0 < M);
  const uint32_t bbase     = lds_addr(&sB[0][0][0]);
  const uint32_t bufstride = 32u * 72u * 2u;       // 4608 B

  auto stageB = [&](int k0s, int buf) {
    // 32 rows x 128 B/row = 256 x 16B segments; 2 per thread
#pragma unroll
    for (int i = 0; i < 2; ++i) {
      const int s = tid + (i << 7);
      const int row = s >> 3, cs = s & 7;
      const uint32_t l_off = bbase + (uint32_t)buf * bufstride +
                             (uint32_t)row * 144u + (uint32_t)cs * 16u;
      const uint32_t g_off =
          (uint32_t)(((size_t)(k0s + row) * N + n0) * 2 + cs * 16);
      async_ld_b128(l_off, g_off, Bm);
    }
  };

  stageB(0, 0);
  v8f acc[4] = {};
  for (int k0 = 0; k0 < K; k0 += 32) {
    wait_async0();
    __syncthreads();
    const int cur = (k0 >> 5) & 1;
    if (k0 + 32 < K) stageB(k0 + 32, cur ^ 1);     // overlaps with WMMAs below
    if (active) {
      v16bf a = load_a_frag(A, K, m0, k0, lane);
#pragma unroll
      for (int j = 0; j < 4; ++j) {
        BF16x16 b;
#pragma unroll
        for (int s = 0; s < 8; ++s) {
          b.v[s]     = sB[cur][((lane >> 4) << 3) + s][j * 16 + (lane & 15)];
          b.v[8 + s] = sB[cur][16 + ((lane >> 4) << 3) + s][j * 16 + (lane & 15)];
        }
        acc[j] = wmma_bf16(a, b.v, acc[j]);
      }
    }
  }
  if (active) {
    const int rbase = m0 + ((lane >> 4) << 3);
#pragma unroll
    for (int j = 0; j < 4; ++j) {
      const int col = n0 + 16 * j + (lane & 15);
      const float bv = bias ? bias[col] : 0.0f;
#pragma unroll
      for (int r = 0; r < 8; ++r) {
        const float v = acc[j][r] + bv;
        const size_t idx = (size_t)(rbase + r) * N + col;
        if (C)   C[idx]   = v;
        if (Cbf) Cbf[idx] = (bf16_t)v;
      }
    }
  }
}

// GEMM with concatenated-K A = [A1 | A2]:  C = [A1,A2] @ B + bias  (one wave/block)
__global__ void k_wmma_gemm_cat(const bf16_t* __restrict__ A1, int K1,
                                const bf16_t* __restrict__ A2, int K2,
                                const bf16_t* __restrict__ Bm, const float* __restrict__ bias,
                                float* __restrict__ C, bf16_t* __restrict__ Cbf,
                                int M, int N) {
  const int lane = threadIdx.x & 31;
  const int n0 = blockIdx.x * 64;
  const int m0 = blockIdx.y * 16;
  v8f acc[4] = {};
  for (int k0 = 0; k0 < K1; k0 += 32) {
    v16bf a = load_a_frag(A1, K1, m0, k0, lane);
#pragma unroll
    for (int j = 0; j < 4; ++j)
      acc[j] = wmma_bf16(a, load_b_frag(Bm, N, k0, n0 + 16 * j, lane), acc[j]);
  }
  for (int k0 = 0; k0 < K2; k0 += 32) {
    v16bf a = load_a_frag(A2, K2, m0, k0, lane);
#pragma unroll
    for (int j = 0; j < 4; ++j)
      acc[j] = wmma_bf16(a, load_b_frag(Bm, N, K1 + k0, n0 + 16 * j, lane), acc[j]);
  }
  const int rbase = m0 + ((lane >> 4) << 3);
#pragma unroll
  for (int j = 0; j < 4; ++j) {
    const int col = n0 + 16 * j + (lane & 15);
    const float bv = bias ? bias[col] : 0.0f;
#pragma unroll
    for (int r = 0; r < 8; ++r) {
      const float v = acc[j][r] + bv;
      const size_t idx = (size_t)(rbase + r) * N + col;
      if (C)   C[idx]   = v;
      if (Cbf) Cbf[idx] = (bf16_t)v;
    }
  }
}

// ---------------- fused LSTM cell step ----------------
// gates = [x, h_prev] @ W(2H x 4H) + bias ; pointwise i/j/f/o ; forget_bias = 1.0
// block = 4 waves; wave w owns 16 gate columns n0 = bx*64 + w*16 (x4 gates).
// All waves share the 32(B)x32(K) activation tile staged in LDS (async, dbl buf);
// each wave register-pipelines its own four W fragments (next-k prefetched).
__global__ __launch_bounds__(128) void k_lstm_cell(
    const bf16_t* __restrict__ x,        // [B,H] bf16
    const bf16_t* __restrict__ hprev,    // [B,H] bf16
    float* __restrict__ c,               // [B,H] in/out (in-place)
    float* __restrict__ h_out_f,         // [B,H] fp32
    bf16_t* __restrict__ h_out_bf,       // [B,H] bf16
    const bf16_t* __restrict__ W,        // [2H,4H] bf16
    const float* __restrict__ bias,      // [4H]
    float* __restrict__ h_copy) {        // optional [B,H] fp32
  __shared__ bf16_t sA[2][32][40];                 // +8 bf16 row pad
  const int tid  = threadIdx.x;
  const int lane = tid & 31;
  const int wv   = tid >> 5;
  const int n0   = blockIdx.x * 64 + wv * 16;
  const int ldw  = 4 * HH;
  const uint32_t abase     = lds_addr(&sA[0][0][0]);
  const uint32_t bufstride = 32u * 40u * 2u;       // 2560 B

  auto stageA = [&](int k, int buf) {
    const bf16_t* src = (k < HH) ? x : hprev;
    const int kk = (k < HH) ? k : (k - HH);
    // 32 rows x 64 B = 128 x 16B segments; 1 per thread
    const int row = tid >> 2, cs = tid & 3;
    const uint32_t l_off = abase + (uint32_t)buf * bufstride +
                           (uint32_t)row * 80u + (uint32_t)cs * 16u;
    const uint32_t g_off = (uint32_t)(((size_t)row * HH + kk) * 2 + cs * 16);
    async_ld_b128(l_off, g_off, src);
  };

  stageA(0, 0);
  v16bf bc[4], bn[4];
#pragma unroll
  for (int g = 0; g < 4; ++g) bc[g] = load_b_frag(W, ldw, 0, n0 + g * HH, lane);

  v8f acc[2][4] = {};
  for (int k = 0; k < 2 * HH; k += 32) {
    wait_async0();
    __syncthreads();
    const int cur = (k >> 5) & 1;
    const bool more = (k + 32 < 2 * HH);
    if (more) {
      stageA(k + 32, cur ^ 1);
#pragma unroll
      for (int g = 0; g < 4; ++g)
        bn[g] = load_b_frag(W, ldw, k + 32, n0 + g * HH, lane);
      __builtin_prefetch((const void*)(W + (size_t)(k + 64) * ldw + n0), 0, 1);
    }
    const int r    = lane & 15;
    const int koff = (lane >> 4) << 3;
    BF16x16 a0, a1;
    a0.h[0] = *(const v8bf*)&sA[cur][r][koff];
    a0.h[1] = *(const v8bf*)&sA[cur][r][16 + koff];
    a1.h[0] = *(const v8bf*)&sA[cur][r + 16][koff];
    a1.h[1] = *(const v8bf*)&sA[cur][r + 16][16 + koff];
#pragma unroll
    for (int g = 0; g < 4; ++g) {
      acc[0][g] = wmma_bf16(a0.v, bc[g], acc[0][g]);
      acc[1][g] = wmma_bf16(a1.v, bc[g], acc[1][g]);
    }
    if (more) {
#pragma unroll
      for (int g = 0; g < 4; ++g) bc[g] = bn[g];
    }
  }

  // pointwise gates
  const int col = n0 + (lane & 15);
  const float bi = bias[col];
  const float bj = bias[col + HH];
  const float bf = bias[col + 2 * HH];
  const float bo = bias[col + 3 * HH];
#pragma unroll
  for (int mt = 0; mt < 2; ++mt) {
    const int rbase = mt * 16 + ((lane >> 4) << 3);
#pragma unroll
    for (int r = 0; r < 8; ++r) {
      const int row = rbase + r;
      const float iv = acc[mt][0][r] + bi;
      const float jv = acc[mt][1][r] + bj;
      const float fv = acc[mt][2][r] + bf;
      const float ov = acc[mt][3][r] + bo;
      const size_t idx = (size_t)row * HH + col;
      const float cn = c[idx] * sigmoidf_(fv + 1.0f) + sigmoidf_(iv) * tanhf(jv);
      const float hn = tanhf(cn) * sigmoidf_(ov);
      c[idx] = cn;
      h_out_f[idx] = hn;
      h_out_bf[idx] = (bf16_t)hn;
      if (h_copy) h_copy[idx] = hn;
    }
  }
}

// ---------------- attention: scores + softmax + context ----------------
__global__ void k_attention(const float* __restrict__ h,       // [B,H]
                            const float* __restrict__ enc_hs,  // [T,B,H]
                            float* __restrict__ ctx_f,         // [B,H]
                            bf16_t* __restrict__ ctx_bf) {     // [B,H]
  const int b = blockIdx.x, tid = threadIdx.x;
  __shared__ float sc[TT];
  if (tid < TT) {
    const float* eh = enc_hs + ((size_t)tid * BB + b) * HH;
    const float* hb = h + (size_t)b * HH;
    float s = 0.0f;
    for (int k = 0; k < HH; ++k) s += hb[k] * eh[k];
    sc[tid] = s;
  }
  __syncthreads();
  if (tid == 0) {
    float m = -1e30f;
    for (int t = 0; t < TT; ++t) m = fmaxf(m, sc[t]);
    float sum = 0.0f;
    for (int t = 0; t < TT; ++t) { sc[t] = __expf(sc[t] - m); sum += sc[t]; }
    const float inv = 1.0f / sum;
    for (int t = 0; t < TT; ++t) sc[t] *= inv;
  }
  __syncthreads();
  for (int hh = tid; hh < HH; hh += blockDim.x) {
    float acc = 0.0f;
    for (int t = 0; t < TT; ++t) acc += sc[t] * enc_hs[((size_t)t * BB + b) * HH + hh];
    ctx_f[(size_t)b * HH + hh] = acc;
    ctx_bf[(size_t)b * HH + hh] = (bf16_t)acc;
  }
}

// ---------------- final vocab softmax, write probs to d_out[b, t, :] ----------------
__global__ void k_softmax_vt(const float* __restrict__ logits,  // [B,VT]
                             float* __restrict__ out,           // [B,T,VT]
                             int t) {
  const int b = blockIdx.x, tid = threadIdx.x;
  __shared__ float red[256];
  const float* row = logits + (size_t)b * VTOK;
  float m = -1e30f;
  for (int v = tid; v < VTOK; v += 256) m = fmaxf(m, row[v]);
  red[tid] = m; __syncthreads();
  for (int s = 128; s > 0; s >>= 1) {
    if (tid < s) red[tid] = fmaxf(red[tid], red[tid + s]);
    __syncthreads();
  }
  m = red[0]; __syncthreads();
  float sum = 0.0f;
  for (int v = tid; v < VTOK; v += 256) sum += __expf(row[v] - m);
  red[tid] = sum; __syncthreads();
  for (int s = 128; s > 0; s >>= 1) {
    if (tid < s) red[tid] += red[tid + s];
    __syncthreads();
  }
  const float inv = 1.0f / red[0];
  float* o = out + ((size_t)b * TT + t) * VTOK;
  for (int v = tid; v < VTOK; v += 256) o[v] = __expf(row[v] - m) * inv;
}

// ---------------- host driver ----------------
static inline void* bump(char*& p, size_t bytes) {
  void* r = (void*)p;
  p += (bytes + 255) & ~(size_t)255;
  return r;
}

extern "C" void kernel_launch(void* const* d_in, const int* in_sizes, int n_in,
                              void* d_out, int out_size, void* d_ws, size_t ws_size,
                              hipStream_t stream) {
  const int*   source    = (const int*)d_in[0];
  const int*   target    = (const int*)d_in[1];
  const float* s_emb     = (const float*)d_in[2];
  const float* s_proj_W  = (const float*)d_in[3];
  const float* s_proj_b  = (const float*)d_in[4];
  const float* t_emb     = (const float*)d_in[5];
  const float* t_proj_W  = (const float*)d_in[6];
  const float* t_proj_b  = (const float*)d_in[7];
  const float* enc_W     = (const float*)d_in[8];
  const float* enc_b     = (const float*)d_in[9];
  const float* dec_W     = (const float*)d_in[10];
  const float* dec_b     = (const float*)d_in[11];
  const float* W_c       = (const float*)d_in[12];
  const float* b_c       = (const float*)d_in[13];
  const float* proj_W    = (const float*)d_in[14];
  const float* proj_b    = (const float*)d_in[15];
  const float* proj_Wo   = (const float*)d_in[16];
  const float* proj_bo   = (const float*)d_in[17];
  float* out = (float*)d_out;

  const size_t lstmW = (size_t)2 * HH * 4 * HH;   // per-layer weight elements
  char* p = (char*)d_ws;
  bf16_t* sW_bf   = (bf16_t*)bump(p, (size_t)EE * HH * 2);
  bf16_t* tW_bf   = (bf16_t*)bump(p, (size_t)EE * HH * 2);
  bf16_t* encW_bf = (bf16_t*)bump(p, (size_t)LL * lstmW * 2);
  bf16_t* decW_bf = (bf16_t*)bump(p, (size_t)LL * lstmW * 2);
  bf16_t* Wc_bf   = (bf16_t*)bump(p, (size_t)2 * HH * HH * 2);
  bf16_t* pW_bf   = (bf16_t*)bump(p, (size_t)HH * EE * 2);
  bf16_t* pWo_bf  = (bf16_t*)bump(p, (size_t)EE * VTOK * 2);
  bf16_t* srcE    = (bf16_t*)bump(p, (size_t)TT * BB * EE * 2);
  bf16_t* tgtE    = (bf16_t*)bump(p, (size_t)TT * BB * EE * 2);
  bf16_t* srcX    = (bf16_t*)bump(p, (size_t)TT * BB * HH * 2);
  bf16_t* tgtX    = (bf16_t*)bump(p, (size_t)TT * BB * HH * 2);
  float*  enc_hs  = (float*) bump(p, (size_t)TT * BB * HH * 4);
  float*  c_enc   = (float*) bump(p, (size_t)LL * BB * HH * 4);
  float*  c_dec   = (float*) bump(p, (size_t)LL * BB * HH * 4);
  bf16_t* hb_enc  = (bf16_t*)bump(p, (size_t)LL * 2 * BB * HH * 2);  // ping-pong
  bf16_t* hb_dec  = (bf16_t*)bump(p, (size_t)LL * 2 * BB * HH * 2);
  float*  hf_enc  = (float*) bump(p, (size_t)LL * 2 * BB * HH * 4);
  float*  hf_dec  = (float*) bump(p, (size_t)LL * 2 * BB * HH * 4);
  float*  ctx_f   = (float*) bump(p, (size_t)BB * HH * 4);
  bf16_t* ctx_bf  = (bf16_t*)bump(p, (size_t)BB * HH * 2);
  bf16_t* ht_bf   = (bf16_t*)bump(p, (size_t)BB * HH * 2);
  bf16_t* oe_bf   = (bf16_t*)bump(p, (size_t)BB * EE * 2);
  float*  logits  = (float*) bump(p, (size_t)BB * VTOK * 4);

  #define HB_ENC(l, pp) (hb_enc + ((size_t)(l) * 2 + (pp)) * BB * HH)
  #define HB_DEC(l, pp) (hb_dec + ((size_t)(l) * 2 + (pp)) * BB * HH)
  #define HF_ENC(l, pp) (hf_enc + ((size_t)(l) * 2 + (pp)) * BB * HH)
  #define HF_DEC(l, pp) (hf_dec + ((size_t)(l) * 2 + (pp)) * BB * HH)

  // zero-init recurrent state (both pings)
  hipMemsetAsync(c_enc,  0, (size_t)LL * BB * HH * 4, stream);
  hipMemsetAsync(c_dec,  0, (size_t)LL * BB * HH * 4, stream);
  hipMemsetAsync(hb_enc, 0, (size_t)LL * 2 * BB * HH * 2, stream);
  hipMemsetAsync(hb_dec, 0, (size_t)LL * 2 * BB * HH * 2, stream);

  // weight dtype conversion (amortized across the 64-step loops)
  k_f32_to_bf16<<<2048, 256, 0, stream>>>(s_proj_W, sW_bf,   (long)EE * HH);
  k_f32_to_bf16<<<2048, 256, 0, stream>>>(t_proj_W, tW_bf,   (long)EE * HH);
  k_f32_to_bf16<<<4096, 256, 0, stream>>>(enc_W,    encW_bf, (long)LL * lstmW);
  k_f32_to_bf16<<<4096, 256, 0, stream>>>(dec_W,    decW_bf, (long)LL * lstmW);
  k_f32_to_bf16<<<2048, 256, 0, stream>>>(W_c,      Wc_bf,   (long)2 * HH * HH);
  k_f32_to_bf16<<<2048, 256, 0, stream>>>(proj_W,   pW_bf,   (long)HH * EE);
  k_f32_to_bf16<<<4096, 256, 0, stream>>>(proj_Wo,  pWo_bf,  (long)EE * VTOK);

  // embedding gather (already time-major [T*B, E])
  k_gather_embed<<<TT * BB, 256, 0, stream>>>(source, s_emb, srcE);
  k_gather_embed<<<TT * BB, 256, 0, stream>>>(target, t_emb, tgtE);

  // batched input projections for all timesteps: [T*B, E] @ [E, H] + b
  k_wmma_gemm<<<dim3(HH / 64, (TT * BB) / 64), 128, 0, stream>>>(
      srcE, sW_bf, s_proj_b, nullptr, srcX, TT * BB, HH, EE);
  k_wmma_gemm<<<dim3(HH / 64, (TT * BB) / 64), 128, 0, stream>>>(
      tgtE, tW_bf, t_proj_b, nullptr, tgtX, TT * BB, HH, EE);

  // ---------------- encoder ----------------
  {
    int p0 = 0, p1 = 0;
    for (int t = 0; t < TT; ++t) {
      k_lstm_cell<<<HH / 64, 128, 0, stream>>>(
          srcX + (size_t)t * BB * HH, HB_ENC(0, p0), c_enc,
          HF_ENC(0, p0 ^ 1), HB_ENC(0, p0 ^ 1),
          encW_bf, enc_b, nullptr);
      k_lstm_cell<<<HH / 64, 128, 0, stream>>>(
          HB_ENC(0, p0 ^ 1), HB_ENC(1, p1), c_enc + (size_t)BB * HH,
          HF_ENC(1, p1 ^ 1), HB_ENC(1, p1 ^ 1),
          encW_bf + lstmW, enc_b + 4 * HH,
          enc_hs + (size_t)t * BB * HH);
      p0 ^= 1; p1 ^= 1;
    }
  }

  // ---------------- decoder ----------------
  {
    int p0 = 0, p1 = 0;
    for (int t = 0; t < TT; ++t) {
      k_lstm_cell<<<HH / 64, 128, 0, stream>>>(
          tgtX + (size_t)t * BB * HH, HB_DEC(0, p0), c_dec,
          HF_DEC(0, p0 ^ 1), HB_DEC(0, p0 ^ 1),
          decW_bf, dec_b, nullptr);
      k_lstm_cell<<<HH / 64, 128, 0, stream>>>(
          HB_DEC(0, p0 ^ 1), HB_DEC(1, p1), c_dec + (size_t)BB * HH,
          HF_DEC(1, p1 ^ 1), HB_DEC(1, p1 ^ 1),
          decW_bf + lstmW, dec_b + 4 * HH, nullptr);

      k_attention<<<BB, 256, 0, stream>>>(HF_DEC(1, p1 ^ 1), enc_hs, ctx_f, ctx_bf);

      // h_t = [h, ctx] @ W_c + b_c        (M=32, N=1024, K=2048)
      k_wmma_gemm_cat<<<dim3(HH / 64, BB / 16), 32, 0, stream>>>(
          HB_DEC(1, p1 ^ 1), HH, ctx_bf, HH, Wc_bf, b_c, nullptr, ht_bf, BB, HH);
      // outemb = h_t @ proj_W + proj_b    (M=32, N=512, K=1024)
      k_wmma_gemm<<<dim3(EE / 64, 1), 128, 0, stream>>>(
          ht_bf, pW_bf, proj_b, nullptr, oe_bf, BB, EE, HH);
      // logits = outemb @ proj_Wo + bo    (M=32, N=32000, K=512)
      k_wmma_gemm<<<dim3(VTOK / 64, 1), 128, 0, stream>>>(
          oe_bf, pWo_bf, proj_bo, logits, nullptr, BB, VTOK, EE);

      k_softmax_vt<<<BB, 256, 0, stream>>>(logits, out, t);
      p0 ^= 1; p1 ^= 1;
    }
  }
  #undef HB_ENC
  #undef HB_DEC
  #undef HF_ENC
  #undef HF_DEC
}
